// Attention_8924942041572
// MI455X (gfx1250) — matrix operand order repaired and checked by reading, not compiled, not run on hardware
//
#include <hip/hip_runtime.h>
#include <hip/hip_bf16.h>
#include <stdint.h>

#define SEQ   2048
#define DIM   4096
#define NH    32
#define NKV   8
#define HD    128
#define KVDIM 1024
#define NEG_INF -1000000000.0f

typedef __attribute__((ext_vector_type(16))) __bf16 v16bf;
typedef __attribute__((ext_vector_type(8)))  float  v8f;

union Frag {
  v16bf    v;
  uint32_t u[8];
  uint4    q[2];
};

// Pack two f32 into packed bf16 (truncate) with one v_perm_b32.
__device__ __forceinline__ uint32_t pk2(float a, float b) {
  return __builtin_amdgcn_perm(__float_as_uint(b), __float_as_uint(a), 0x07060302u);
}
__device__ __forceinline__ uint16_t f2b(float a) { return (uint16_t)(__float_as_uint(a) >> 16); }
__device__ __forceinline__ float    b2f(uint16_t h) { return __uint_as_float(((uint32_t)h) << 16); }

// Cross-half exchange (lane <-> lane^16) as a VALU op: v_permlanex16_b32.
__device__ __forceinline__ uint32_t xhalf_u(uint32_t x) {
  return (uint32_t)__builtin_amdgcn_permlanex16((int)x, (int)x, 0x76543210, 0xFEDCBA98, false, false);
}
__device__ __forceinline__ float xhalf_f(float x) {
  return __int_as_float(__builtin_amdgcn_permlanex16(__float_as_int(x), __float_as_int(x),
                                                     0x76543210, 0xFEDCBA98, false, false));
}

__device__ __forceinline__ v8f wmma_bf16(const Frag& a, const Frag& b, v8f c) {
  return __builtin_amdgcn_wmma_f32_16x16x32_bf16(false, a.v, false, b.v, (short)0, c, false, false);
}

// A fragment (16x32 bf16, M x K): lane = row (lane&15); lanes<16 hold K 0..15,
// lanes>=16 hold K 16..31; two b128 loads.
__device__ __forceinline__ Frag load_a_bf16(const uint16_t* rowbase, int lane) {
  Frag f;
  const uint16_t* p = rowbase + ((lane >> 4) << 4);
  f.q[0] = *(const uint4*)(p);
  f.q[1] = *(const uint4*)(p + 8);
  return f;
}
// B fragment (32x16 bf16, K x N), source with K contiguous per column:
// lane = column (lane&15); lanes<16 hold K 0..15, lanes>=16 hold K 16..31.
__device__ __forceinline__ Frag load_b_bf16(const uint16_t* colbase, int lane) {
  Frag f;
  const uint16_t* p = colbase + ((lane >> 4) << 4);
  f.q[0] = *(const uint4*)(p);
  f.q[1] = *(const uint4*)(p + 8);
  return f;
}
// B fragment from fp32 [N,K] weights, packed to bf16 inline (8x v_perm).
__device__ __forceinline__ Frag load_b_f32(const float* rowbase, int lane) {
  Frag f;
  const float* p = rowbase + ((lane >> 4) << 4);
  float4 a = *(const float4*)(p);
  float4 b = *(const float4*)(p + 4);
  float4 c = *(const float4*)(p + 8);
  float4 d = *(const float4*)(p + 12);
  f.u[0] = pk2(a.x, a.y); f.u[1] = pk2(a.z, a.w);
  f.u[2] = pk2(b.x, b.y); f.u[3] = pk2(b.z, b.w);
  f.u[4] = pk2(c.x, c.y); f.u[5] = pk2(c.z, c.w);
  f.u[6] = pk2(d.x, d.y); f.u[7] = pk2(d.z, d.w);
  return f;
}

__global__ void cvt_kernel(const float* __restrict__ in, uint16_t* __restrict__ out, int n) {
  int i = blockIdx.x * blockDim.x + threadIdx.x;
  if (i < n) out[i] = f2b(in[i]);
}

// Y = A[M,K](bf16) @ W[N,K](f32)^T ; per-wave 32x64 tile, K stepped by 32.
// MODE: 0 = f32 row-major, 1 = bf16 row-major, 2 = bf16 transposed (out[n*M + m]).
template<int MODE>
__global__ __launch_bounds__(256) void gemm_xwt(const uint16_t* __restrict__ A,
                                                const float* __restrict__ W,
                                                void* __restrict__ out,
                                                int M, int N, int K) {
  const int lane = threadIdx.x & 31;
  const int wave = threadIdx.x >> 5;
  const int tilesN = N >> 6;
  const int wid = blockIdx.x * 8 + wave;
  const int mt = wid / tilesN, nt = wid % tilesN;
  if (mt >= (M >> 5)) return;
  const int m0 = mt << 5, n0 = nt << 6;

  v8f acc[2][4] = {};
  const uint16_t* arow0 = A + (size_t)(m0 + (lane & 15)) * K;
  const uint16_t* arow1 = arow0 + (size_t)16 * K;
  const float*    wrow  = W + (size_t)(n0 + (lane & 15)) * K;

  for (int kk = 0; kk < K; kk += 32) {
    if (kk + 256 < K) {
      __builtin_prefetch(arow0 + kk + 256, 0, 0);
      __builtin_prefetch(wrow + kk + 256, 0, 0);
    }
    Frag a0 = load_a_bf16(arow0 + kk, lane);
    Frag a1 = load_a_bf16(arow1 + kk, lane);
#pragma unroll
    for (int f = 0; f < 4; ++f) {
      Frag bf = load_b_f32(wrow + (size_t)(f << 4) * K + kk, lane);
      acc[0][f] = wmma_bf16(a0, bf, acc[0][f]);
      acc[1][f] = wmma_bf16(a1, bf, acc[1][f]);
    }
  }
  const int ncol = lane & 15, rowoff = (lane >> 4) << 3;
#pragma unroll
  for (int h = 0; h < 2; ++h)
#pragma unroll
    for (int f = 0; f < 4; ++f)
#pragma unroll
      for (int r = 0; r < 8; ++r) {
        int m = m0 + h * 16 + r + rowoff;
        int n = n0 + (f << 4) + ncol;
        if (MODE == 0) ((float*)out)[(size_t)m * N + n]    = acc[h][f][r];
        if (MODE == 1) ((uint16_t*)out)[(size_t)m * N + n] = f2b(acc[h][f][r]);
        if (MODE == 2) ((uint16_t*)out)[(size_t)n * M + m] = f2b(acc[h][f][r]);
      }
}

__global__ void rope_kernel(uint16_t* __restrict__ t, int heads,
                            const float* __restrict__ cs, const float* __restrict__ sn) {
  int idx = blockIdx.x * blockDim.x + threadIdx.x;
  if (idx >= SEQ * heads * 64) return;
  int i   = idx & 63;
  int h   = (idx >> 6) % heads;
  int pos = idx / (heads * 64);
  size_t base = (size_t)pos * heads * HD + (size_t)h * HD + 2 * i;
  float re = b2f(t[base]), im = b2f(t[base + 1]);
  float c = cs[pos * 64 + i], s = sn[pos * 64 + i];
  t[base]     = f2b(re * c - im * s);
  t[base + 1] = f2b(re * s + im * c);
}

// Flash attention, transposed-score formulation (no LDS, no ds traffic):
//   S^T = K_tile(16x d) x Q^T   -> kv index on VGPR axis, query index on lane axis
//   O^T = V^T x P^T             -> V^T straight from pre-transposed Vt[d][seq]
// One wave owns 16 query columns of one head; kv tiles of 32.
__global__ __launch_bounds__(256) void attn_kernel(const uint16_t* __restrict__ Q,
                                                   const uint16_t* __restrict__ Kc,
                                                   const uint16_t* __restrict__ Vt,
                                                   uint16_t* __restrict__ O) {
  const int lane  = threadIdx.x & 31;
  const int wave  = threadIdx.x >> 5;
  const int wid   = blockIdx.x * 8 + wave;
  const int head  = wid >> 7;
  const int q0    = (wid & 127) << 4;
  const int kvh   = head >> 2;
  const int ncol  = lane & 15;       // query column owned by this lane
  const int g     = (lane >> 4) & 1; // half-wave group
  const float scale = 0.08838834764831845f; // 1/sqrt(128)

  // Q^T as B fragments (resident): one per 32-wide d chunk.
  Frag qb[4];
#pragma unroll
  for (int dd = 0; dd < 4; ++dd)
    qb[dd] = load_b_bf16(Q + (size_t)(q0 + ncol) * DIM + head * HD + dd * 32, lane);

  const int mcol = q0 + ncol;  // global query index for this lane
  float mst = -1e30f, lst = 0.f;
  v8f oacc[8] = {};            // O^T: 8 frags over d (rows), query on lanes

  for (int n0 = 0; n0 <= q0 + 15; n0 += 32) {
    // S^T tile (32 kv rows x 16 query cols) as two C fragments.
    v8f s0 = {}, s1 = {};
#pragma unroll
    for (int dd = 0; dd < 4; ++dd) {
      Frag ka0 = load_a_bf16(Kc + (size_t)(n0 + ncol) * KVDIM + kvh * HD + dd * 32, lane);
      Frag ka1 = load_a_bf16(Kc + (size_t)(n0 + 16 + ncol) * KVDIM + kvh * HD + dd * 32, lane);
      s0 = wmma_bf16(ka0, qb[dd], s0);
      s1 = wmma_bf16(ka1, qb[dd], s1);
    }
    // Scale + causal mask (kv index n on the register axis, per-lane query col).
    const bool diag = (n0 + 31 > q0);
    float e0[8], e1[8];
#pragma unroll
    for (int r = 0; r < 8; ++r) {
      int n = n0 + r + (g << 3);
      float a = s0[r] * scale;
      float b = s1[r] * scale;
      if (diag) {
        if (n > mcol)      a += NEG_INF;
        if (n + 16 > mcol) b += NEG_INF;
      }
      e0[r] = a; e1[r] = b;
    }
    // Row (over kv) max: per-lane over 16 values, then one cross-half exchange.
    float t = e0[0];
#pragma unroll
    for (int r = 1; r < 8; ++r) t = fmaxf(t, e0[r]);
#pragma unroll
    for (int r = 0; r < 8; ++r) t = fmaxf(t, e1[r]);
    t = fmaxf(t, xhalf_f(t));
    float mnew  = fmaxf(mst, t);
    float alpha = __expf(mst - mnew);
    mst = mnew;
    float srow = 0.f;
#pragma unroll
    for (int r = 0; r < 8; ++r) { e0[r] = __expf(e0[r] - mnew); srow += e0[r]; }
#pragma unroll
    for (int r = 0; r < 8; ++r) { e1[r] = __expf(e1[r] - mnew); srow += e1[r]; }
    srow += xhalf_f(srow);
    lst = lst * alpha + srow;
#pragma unroll
    for (int c = 0; c < 8; ++c)
#pragma unroll
      for (int r = 0; r < 8; ++r) oacc[c][r] *= alpha;

    // Build P^T B-fragment in registers: pack pairs, exchange halves.
    uint32_t p0[4], p1[4];
#pragma unroll
    for (int i = 0; i < 4; ++i) {
      p0[i] = pk2(e0[2 * i], e0[2 * i + 1]);
      p1[i] = pk2(e1[2 * i], e1[2 * i + 1]);
    }
    Frag pb;
#pragma unroll
    for (int i = 0; i < 4; ++i) {
      uint32_t x0 = xhalf_u(p0[i]);  // partner's s0 pairs
      uint32_t x1 = xhalf_u(p1[i]);  // partner's s1 pairs
      pb.u[i]     = g ? x1 : p0[i];  // k = n0 + {0..7}  | {16..23}
      pb.u[i + 4] = g ? p1[i] : x0;  // k = n0 + {8..15} | {24..31}
    }
    // O^T += V^T x P^T ; V^T fragments are contiguous b128 loads from Vt.
#pragma unroll
    for (int c = 0; c < 8; ++c) {
      Frag va = load_a_bf16(Vt + (size_t)(kvh * HD + (c << 4) + ncol) * SEQ + n0, lane);
      oacc[c] = wmma_bf16(va, pb, oacc[c]);
    }
  }
  // Epilogue: divide by row sum, pack 8 consecutive d per fragment -> one b128 store.
  float inv = 1.0f / lst;
  uint16_t* obase = O + (size_t)mcol * DIM + head * HD + (g << 3);
#pragma unroll
  for (int c = 0; c < 8; ++c) {
    uint4 w;
    w.x = pk2(oacc[c][0] * inv, oacc[c][1] * inv);
    w.y = pk2(oacc[c][2] * inv, oacc[c][3] * inv);
    w.z = pk2(oacc[c][4] * inv, oacc[c][5] * inv);
    w.w = pk2(oacc[c][6] * inv, oacc[c][7] * inv);
    *(uint4*)(obase + (c << 4)) = w;
  }
}

extern "C" void kernel_launch(void* const* d_in, const int* in_sizes, int n_in,
                              void* d_out, int out_size, void* d_ws, size_t ws_size,
                              hipStream_t stream) {
  const float* x  = (const float*)d_in[0];
  const float* wq = (const float*)d_in[1];
  const float* wk = (const float*)d_in[2];
  const float* wv = (const float*)d_in[3];
  const float* wo = (const float*)d_in[4];
  const float* fc = (const float*)d_in[5];
  const float* fs = (const float*)d_in[6];
  // d_in[7] = dense mask (causal computed in-kernel), d_in[8] = start_pos (== 0).

  uint8_t* p = (uint8_t*)d_ws;
  uint16_t* xb = (uint16_t*)p; p += (size_t)SEQ * DIM * 2;
  uint16_t* qb = (uint16_t*)p; p += (size_t)SEQ * DIM * 2;
  uint16_t* kb = (uint16_t*)p; p += (size_t)SEQ * KVDIM * 2;
  uint16_t* vt = (uint16_t*)p; p += (size_t)SEQ * KVDIM * 2;  // transposed [KVDIM][SEQ]
  uint16_t* ab = (uint16_t*)p; p += (size_t)SEQ * DIM * 2;

  int n = SEQ * DIM;
  cvt_kernel<<<(n + 255) / 256, 256, 0, stream>>>(x, xb, n);

  gemm_xwt<1><<<(SEQ / 32) * (DIM / 64) / 8,   256, 0, stream>>>(xb, wq, qb, SEQ, DIM,   DIM);
  gemm_xwt<1><<<(SEQ / 32) * (KVDIM / 64) / 8, 256, 0, stream>>>(xb, wk, kb, SEQ, KVDIM, DIM);
  gemm_xwt<2><<<(SEQ / 32) * (KVDIM / 64) / 8, 256, 0, stream>>>(xb, wv, vt, SEQ, KVDIM, DIM);

  rope_kernel<<<(SEQ * NH  * 64) / 256, 256, 0, stream>>>(qb, NH,  fc, fs);
  rope_kernel<<<(SEQ * NKV * 64) / 256, 256, 0, stream>>>(kb, NKV, fc, fs);

  attn_kernel<<<(NH * (SEQ / 16)) / 8, 256, 0, stream>>>(qb, kb, vt, ab);

  gemm_xwt<0><<<(SEQ / 32) * (DIM / 64) / 8, 256, 0, stream>>>(ab, wo, d_out, SEQ, DIM, DIM);
}
